// causal_kMemoryHMVAE_toeplitz_34703335752114
// MI455X (gfx1250) — matrix operand order repaired and checked by reading, not compiled, not run on hardware
//
#include <hip/hip_runtime.h>
#include <hip/hip_bf16.h>
#include <math.h>

// ---------- types ----------
typedef __attribute__((ext_vector_type(4)))  float          v4f;
typedef __attribute__((ext_vector_type(8)))  float          v8f;
typedef __attribute__((ext_vector_type(16))) __bf16         v16bf;
typedef __attribute__((ext_vector_type(8)))  unsigned short v8us;
typedef __attribute__((ext_vector_type(16))) unsigned short v16us;

union Frag16 {
    v8us  h[2];
    v16us w;
    v16bf bf;
};

__device__ __forceinline__ unsigned short f2bf(float f) {
    unsigned u = __float_as_uint(f);
    unsigned r = (u + 0x7FFFu + ((u >> 16) & 1u)) >> 16;  // RNE truncate
    return (unsigned short)r;
}
__device__ __forceinline__ unsigned pack2bf(float lo, float hi) {
    return (unsigned)f2bf(lo) | ((unsigned)f2bf(hi) << 16);
}

// ---------- small utility kernels ----------
__global__ void zero_kernel(float* __restrict__ p, int n) {
    int i = blockIdx.x * 256 + threadIdx.x;
    if (i < n) p[i] = 0.0f;
}

// W [N x K] fp32 row-major -> Wt bf16 in WMMA-B fragment-major layout:
//   Wt[((kt*NT + nt)*32 + lane)*16 + e],  NT = Npad/16
__global__ void conv_weight_frag_kernel(const float* __restrict__ W, int N, int K,
                                        unsigned short* __restrict__ Wt, int Npad, int Kpad) {
    int idx = blockIdx.x * 256 + threadIdx.x;
    if (idx >= Npad * Kpad) return;
    int e    = idx & 15;
    int lane = (idx >> 4) & 31;
    int t    = idx >> 9;  // tile = kt*NT + nt
    int NT   = Npad >> 4;
    int kt   = t / NT;
    int nt   = t - kt * NT;
    int k    = (kt << 5) + ((lane >> 4) << 4) + e;
    int n    = (nt << 4) + (lane & 15);
    float v  = (k < K && n < N) ? W[n * K + k] : 0.0f;
    Wt[idx] = f2bf(v);
}

// BN finalize: scale = g * rsqrt(var+eps), shift = beta - mean*scale
__global__ void bn_finalize_kernel(const float* __restrict__ sum, const float* __restrict__ sumsq,
                                   const float* __restrict__ g, const float* __restrict__ be,
                                   float* __restrict__ scale, float* __restrict__ shift,
                                   int Nact, int Npad, float invB) {
    int n = blockIdx.x * 64 + threadIdx.x;
    if (n >= Npad) return;
    if (n < Nact) {
        float m  = sum[n] * invB;
        float v  = sumsq[n] * invB - m * m;
        float sc = g[n] * rsqrtf(v + 1e-5f);
        scale[n] = sc;
        shift[n] = be[n] - m * sc;
    } else {
        scale[n] = 0.0f;
        shift[n] = 0.0f;
    }
}

// ---------- WMMA GEMM:  Y = [relu](BN(X) @ Wt + bias), optional column stats ----------
// Block tile 64(M) x 64(N); 8 waves = 4 M-subtiles x 2 wave-columns; each wave holds one
// A fragment + two B fragments -> 2 x v_wmma_f32_16x16x32_bf16 per 32-K step.
// Requires Brows % 64 == 0 (grid.x = Brows/64).
template <bool RELU, bool STATS, bool AFFINE, bool NTSTORE>
__global__ __launch_bounds__(256) void gemm_bf16_wmma(
        const float* __restrict__ X, int ldx, int Kpad,
        const unsigned short* __restrict__ Wt, int Npad, int Nact,
        const float* __restrict__ bias,
        const float* __restrict__ scale, const float* __restrict__ shift,
        float* __restrict__ Y, int ldy,
        float* __restrict__ colsum, float* __restrict__ colsumsq) {
    __shared__ alignas(16) unsigned short As[64][40];  // 64 rows x 32 K bf16, 80B row stride

    const int tid  = threadIdx.x;
    const int lane = tid & 31;
    const int wv   = tid >> 5;
    const int wm   = wv & 3;    // M sub-tile 0..3
    const int wp   = wv >> 2;   // wave column 0..1
    const int mb0  = blockIdx.x * 64;
    const int NT   = Npad >> 4;
    const int nt0  = blockIdx.y * 4 + wp * 2;  // first of two 16-wide N sub-tiles

    v8f acc0 = {0.f, 0.f, 0.f, 0.f, 0.f, 0.f, 0.f, 0.f};
    v8f acc1 = {0.f, 0.f, 0.f, 0.f, 0.f, 0.f, 0.f, 0.f};

    // A-staging coordinates (2 x float4 per thread)
    const int r0 = tid >> 3;             // row for slot 0 (0..31)
    const int c0 = (tid & 7) << 2;       // col 0,4,..,28
    const float* xrow0 = X + (size_t)(mb0 + r0) * ldx + c0;
    const float* xrow1 = X + (size_t)(mb0 + r0 + 32) * ldx + c0;

    const int KT = Kpad >> 5;
    for (int ktile = 0; ktile < KT; ++ktile) {
        const int kbase = ktile << 5;
        // --- stage A tile 64x32: v4f loads, affine, pack, ds_store_b64
        {
            const v4f xv0 = *(const v4f*)(xrow0 + kbase);
            const v4f xv1 = *(const v4f*)(xrow1 + kbase);
            v4f s4, h4;
            if (AFFINE) {
                s4 = *(const v4f*)&scale[kbase + c0];
                h4 = *(const v4f*)&shift[kbase + c0];
            }
            float a0_ = xv0.x, b0_ = xv0.y, c0_ = xv0.z, d0_ = xv0.w;
            float a1_ = xv1.x, b1_ = xv1.y, c1_ = xv1.z, d1_ = xv1.w;
            if (AFFINE) {
                a0_ = a0_ * s4.x + h4.x; b0_ = b0_ * s4.y + h4.y;
                c0_ = c0_ * s4.z + h4.z; d0_ = d0_ * s4.w + h4.w;
                a1_ = a1_ * s4.x + h4.x; b1_ = b1_ * s4.y + h4.y;
                c1_ = c1_ * s4.z + h4.z; d1_ = d1_ * s4.w + h4.w;
            }
            uint2 p0, p1;
            p0.x = pack2bf(a0_, b0_); p0.y = pack2bf(c0_, d0_);
            p1.x = pack2bf(a1_, b1_); p1.y = pack2bf(c1_, d1_);
            *(uint2*)&As[r0][c0]      = p0;
            *(uint2*)&As[r0 + 32][c0] = p1;
        }
        __syncthreads();

        // --- A fragment (16x32): lanes 0-15 K{0..15}; lanes 16-31 K{16..31}
        const int arr = wm * 16 + (lane & 15);
        const int ak  = (lane >> 4) * 16;
        Frag16 a;
        a.h[0] = *(const v8us*)&As[arr][ak];
        a.h[1] = *(const v8us*)&As[arr][ak + 8];

        // --- B fragments straight from global (fragment-major layout)
        const unsigned short* bb =
            Wt + ((((size_t)ktile * NT + nt0) << 5) + lane) * 16;
        Frag16 b0, b1;
        b0.h[0] = *(const v8us*)(bb);
        b0.h[1] = *(const v8us*)(bb + 8);
        b1.h[0] = *(const v8us*)(bb + 512);
        b1.h[1] = *(const v8us*)(bb + 520);

        acc0 = __builtin_amdgcn_wmma_f32_16x16x32_bf16(
            false, a.bf, false, b0.bf, (short)0, acc0, false, false);
        acc1 = __builtin_amdgcn_wmma_f32_16x16x32_bf16(
            false, a.bf, false, b1.bf, (short)0, acc1, false, false);
        __syncthreads();
    }

    // --- epilogue: C layout lane n=lane&15, m = 8*(lane>>4) + vgpr
    const int mbase = mb0 + wm * 16 + (lane >> 4) * 8;
    v8f accs[2] = {acc0, acc1};
#pragma unroll
    for (int p = 0; p < 2; ++p) {
        const int n      = ((nt0 + p) << 4) + (lane & 15);
        const bool valid = (n < Nact);
        const int nc     = valid ? n : (Nact - 1);
        const float bv   = bias[nc];  // unconditional load + cndmask (no exec juggling)
        float* yp = Y + (size_t)mbase * ldy + n;
        float s = 0.0f, s2 = 0.0f;
#pragma unroll
        for (int r = 0; r < 8; ++r) {
            float v = accs[p][r] + bv;
            if (RELU) v = fmaxf(v, 0.0f);
            v = valid ? v : 0.0f;  // padded columns stay exact zeros
            if (NTSTORE) __builtin_nontemporal_store(v, yp);
            else         *yp = v;
            yp += ldy;
            if (STATS) { s += v; s2 += v * v; }
        }
        if (STATS) {
            s  += __shfl_xor(s, 16, 32);   // lanes l, l^16 share column n
            s2 += __shfl_xor(s2, 16, 32);
            if ((lane < 16) && valid) {
                atomicAdd(&colsum[n], s);
                atomicAdd(&colsumsq[n], s2);
            }
        }
    }
}

// ---------- alpha_0 head: Linear(150 -> 1), alpha0 = min(exp(.), 500) ----------
__global__ void a0_head_kernel(const float* __restrict__ act, const float* __restrict__ scale,
                               const float* __restrict__ shift, const float* __restrict__ w2,
                               const float* __restrict__ b2, float* __restrict__ alpha0,
                               int Brows, int lda) {
    const int lane = threadIdx.x & 31;
    const int wv   = threadIdx.x >> 5;
    const int row  = blockIdx.x * 8 + wv;
    if (row >= Brows) return;
    float acc = 0.0f;
    for (int k = lane; k < 150; k += 32) {
        float x = act[(size_t)row * lda + k] * scale[k] + shift[k];
        acc += w2[k] * x;
    }
#pragma unroll
    for (int o = 16; o > 0; o >>= 1) acc += __shfl_down(acc, o, 32);
    if (lane == 0) alpha0[row] = fminf(__expf(acc + b2[0]), 500.0f);
}

// ---------- Toeplitz: B[i][j]=Alpha[i-j] (i>=j); C[i][j]=conj(Alpha[32-(i-j)]) (i>j)
// v4f NT stores (write-once 536 MB stream; keep it out of L2).
__global__ __launch_bounds__(256) void toeplitz_kernel(const float* __restrict__ ar_raw,
                                                       const float* __restrict__ alpha0,
                                                       float* __restrict__ outB,
                                                       float* __restrict__ outC) {
    const int b = blockIdx.x;
    __shared__ float Ar[32], Ai[32];
    __shared__ float a0s;
    const int tid = threadIdx.x;
    if (tid == 0) a0s = alpha0[b];
    __syncthreads();
    if (tid < 32) {
        if (tid == 0) {
            Ar[0] = a0s;
            Ai[0] = 0.0f;
        } else {
            float c = 0.022f * a0s;
            Ar[tid] = c * tanhf(ar_raw[(size_t)b * 64 + (tid - 1)]);
            Ai[tid] = c * tanhf(ar_raw[(size_t)b * 64 + (tid + 30)]);
        }
    }
    __syncthreads();
    v4f* Bm = (v4f*)(outB + (size_t)b * 2048);  // 512 v4f per matrix
    v4f* Cm = (v4f*)(outC + (size_t)b * 2048);
#pragma unroll
    for (int e = 0; e < 2; ++e) {
        int idx = tid + e * 256;         // 0..511
        int i  = idx >> 4;               // row 0..31
        int j0 = (idx & 15) << 1;        // col pair
        int d0 = i - j0;
        int d1 = d0 - 1;
        int c0 = d0 < 0 ? 0 : d0, c1 = d1 < 0 ? 0 : d1;
        int e0 = d0 < 1 ? 1 : (32 - d0), e1 = d1 < 1 ? 1 : (32 - d1);
        v4f bv, cv;
        bv.x = (d0 >= 0) ? Ar[c0] : 0.0f;  bv.y = (d0 >= 0) ? Ai[c0] : 0.0f;
        bv.z = (d1 >= 0) ? Ar[c1] : 0.0f;  bv.w = (d1 >= 0) ? Ai[c1] : 0.0f;
        cv.x = (d0 >= 1) ? Ar[e0] : 0.0f;  cv.y = (d0 >= 1) ? -Ai[e0] : 0.0f;
        cv.z = (d1 >= 1) ? Ar[e1] : 0.0f;  cv.w = (d1 >= 1) ? -Ai[e1] : 0.0f;
        __builtin_nontemporal_store(bv, &Bm[idx]);
        __builtin_nontemporal_store(cv, &Cm[idx]);
    }
}

// =======================================================================================
extern "C" void kernel_launch(void* const* d_in, const int* in_sizes, int n_in,
                              void* d_out, int out_size, void* d_ws, size_t ws_size,
                              hipStream_t stream) {
    const int B = in_sizes[0] / 96;  // 32768; multiple of 64 by construction

    const float* z       = (const float*)d_in[0];
    const float* net_w1  = (const float*)d_in[1];
    const float* net_b1  = (const float*)d_in[2];
    const float* net_g1  = (const float*)d_in[3];
    const float* net_be1 = (const float*)d_in[4];
    const float* net_w2  = (const float*)d_in[5];
    const float* net_b2  = (const float*)d_in[6];
    const float* net_g2  = (const float*)d_in[7];
    const float* net_be2 = (const float*)d_in[8];
    const float* mu_w1   = (const float*)d_in[9];
    const float* mu_b1   = (const float*)d_in[10];
    const float* mu_g1   = (const float*)d_in[11];
    const float* mu_be1  = (const float*)d_in[12];
    const float* mu_w2   = (const float*)d_in[13];
    const float* mu_b2   = (const float*)d_in[14];
    const float* a0_w1   = (const float*)d_in[15];
    const float* a0_b1   = (const float*)d_in[16];
    const float* a0_g1   = (const float*)d_in[17];
    const float* a0_be1  = (const float*)d_in[18];
    const float* a0_w2   = (const float*)d_in[19];
    const float* a0_b2   = (const float*)d_in[20];
    const float* ar_w1   = (const float*)d_in[21];
    const float* ar_b1   = (const float*)d_in[22];
    const float* ar_g1   = (const float*)d_in[23];
    const float* ar_be1  = (const float*)d_in[24];
    const float* ar_w2   = (const float*)d_in[25];
    const float* ar_b2   = (const float*)d_in[26];

    // padded dims: N multiples of 64 (block N tile), K multiples of 32
    const int K0 = 96;
    const int P1 = 256, N1 = 200;   // net layer 1
    const int P2 = 448, N2 = 400;   // net layer 2 (h)
    const int P3 = 448, N3 = 400;   // mu hidden
    const int P4 = 192, N4 = 150;   // a0 hidden
    const int P5 = 320, N5 = 300;   // ar hidden
    const int PMU = 256, NMU = 256; // mu out
    const int PAR = 64,  NAR = 62;  // ar out

    // ---- workspace layout (floats; every region 16B aligned) ----
    float* ws = (float*)d_ws;
    size_t off = 0;
    float* bufA  = ws + off; off += (size_t)B * P1;   // act1, later a0 hidden (B*192)
    float* bufB  = ws + off; off += (size_t)B * P2;   // h (persists for 3 heads; ~59MB, L2-resident)
    float* bufC  = ws + off; off += (size_t)B * P3;   // mu hidden, later ar hidden (B*320)
    float* arraw = ws + off; off += (size_t)B * PAR;  // ar linear output
    float* alph0 = ws + off; off += (size_t)B;        // alpha_0
    const int statN = 4 * (P1 + P2 + P3 + P4 + P5);   // [sum,sumsq,scale,shift] per layer
    float* stats = ws + off; off += (size_t)statN;
    float* st1 = stats;
    float* st2 = st1 + 4 * P1;
    float* st3 = st2 + 4 * P2;
    float* st4 = st3 + 4 * P3;
    float* st5 = st4 + 4 * P4;
    unsigned short* w1t   = (unsigned short*)(ws + off);
    unsigned short* w2t   = w1t   + K0 * P1;
    unsigned short* muw1t = w2t   + P1 * P2;
    unsigned short* muw2t = muw1t + P2 * P3;
    unsigned short* a0w1t = muw2t + P3 * PMU;
    unsigned short* arw1t = a0w1t + P2 * P4;
    unsigned short* arw2t = arw1t + P2 * P5;

    float* outMu = (float*)d_out;
    float* outB_ = outMu + (size_t)B * 256;
    float* outC_ = outB_ + (size_t)B * 2048;

    const float invB = 1.0f / (float)B;
    const int gx = B / 64;

    zero_kernel<<<(statN + 255) / 256, 256, 0, stream>>>(stats, statN);

    auto cvt = [&](const float* W, int N, int K, unsigned short* Wt, int Npad, int Kpad) {
        int n = Npad * Kpad;
        conv_weight_frag_kernel<<<(n + 255) / 256, 256, 0, stream>>>(W, N, K, Wt, Npad, Kpad);
    };
    cvt(net_w1, N1, 96,  w1t,   P1, K0);
    cvt(net_w2, N2, 200, w2t,   P2, P1);
    cvt(mu_w1,  N3, 400, muw1t, P3, P2);
    cvt(mu_w2,  NMU, 400, muw2t, PMU, P3);
    cvt(a0_w1,  N4, 400, a0w1t, P4, P2);
    cvt(ar_w1,  N5, 400, arw1t, P5, P2);
    cvt(ar_w2,  NAR, 300, arw2t, PAR, P5);

    // trunk layer 1
    gemm_bf16_wmma<true, true, false, false><<<dim3(gx, P1 / 64), 256, 0, stream>>>(
        z, K0, K0, w1t, P1, N1, net_b1, nullptr, nullptr, bufA, P1,
        st1, st1 + P1);
    bn_finalize_kernel<<<(P1 + 63) / 64, 64, 0, stream>>>(
        st1, st1 + P1, net_g1, net_be1, st1 + 2 * P1, st1 + 3 * P1, N1, P1, invB);

    // trunk layer 2 -> h
    gemm_bf16_wmma<true, true, true, false><<<dim3(gx, P2 / 64), 256, 0, stream>>>(
        bufA, P1, P1, w2t, P2, N2, net_b2, st1 + 2 * P1, st1 + 3 * P1, bufB, P2,
        st2, st2 + P2);
    bn_finalize_kernel<<<(P2 + 63) / 64, 64, 0, stream>>>(
        st2, st2 + P2, net_g2, net_be2, st2 + 2 * P2, st2 + 3 * P2, N2, P2, invB);

    // mu hidden + mu out
    gemm_bf16_wmma<true, true, true, false><<<dim3(gx, P3 / 64), 256, 0, stream>>>(
        bufB, P2, P2, muw1t, P3, N3, mu_b1, st2 + 2 * P2, st2 + 3 * P2, bufC, P3,
        st3, st3 + P3);
    bn_finalize_kernel<<<(P3 + 63) / 64, 64, 0, stream>>>(
        st3, st3 + P3, mu_g1, mu_be1, st3 + 2 * P3, st3 + 3 * P3, N3, P3, invB);
    gemm_bf16_wmma<false, false, true, true><<<dim3(gx, PMU / 64), 256, 0, stream>>>(
        bufC, P3, P3, muw2t, PMU, NMU, mu_b2, st3 + 2 * P3, st3 + 3 * P3, outMu, PMU,
        nullptr, nullptr);

    // alpha_0 head (hidden reuses bufA)
    gemm_bf16_wmma<true, true, true, false><<<dim3(gx, P4 / 64), 256, 0, stream>>>(
        bufB, P2, P2, a0w1t, P4, N4, a0_b1, st2 + 2 * P2, st2 + 3 * P2, bufA, P4,
        st4, st4 + P4);
    bn_finalize_kernel<<<(P4 + 63) / 64, 64, 0, stream>>>(
        st4, st4 + P4, a0_g1, a0_be1, st4 + 2 * P4, st4 + 3 * P4, N4, P4, invB);
    a0_head_kernel<<<(B + 7) / 8, 256, 0, stream>>>(
        bufA, st4 + 2 * P4, st4 + 3 * P4, a0_w2, a0_b2, alph0, B, P4);

    // alpha_rest head (hidden reuses bufC)
    gemm_bf16_wmma<true, true, true, false><<<dim3(gx, P5 / 64), 256, 0, stream>>>(
        bufB, P2, P2, arw1t, P5, N5, ar_b1, st2 + 2 * P2, st2 + 3 * P2, bufC, P5,
        st5, st5 + P5);
    bn_finalize_kernel<<<(P5 + 63) / 64, 64, 0, stream>>>(
        st5, st5 + P5, ar_g1, ar_be1, st5 + 2 * P5, st5 + 3 * P5, N5, P5, invB);
    gemm_bf16_wmma<false, false, true, false><<<dim3(gx, PAR / 64), 256, 0, stream>>>(
        bufC, P5, P5, arw2t, PAR, NAR, ar_b2, st5 + 2 * P5, st5 + 3 * P5, arraw, PAR,
        nullptr, nullptr);

    // Toeplitz expansion (~536 MB of non-temporal v4f stores; the bandwidth floor)
    toeplitz_kernel<<<B, 256, 0, stream>>>(arraw, alph0, outB_, outC_);
}
